// Mutual_Attn_43198781063931
// MI455X (gfx1250) — matrix operand order repaired and checked by reading, not compiled, not run on hardware
//
#include <hip/hip_runtime.h>
#include <hip/hip_bf16.h>

// ---------------------------------------------------------------------------
// Mutual attention (3 segments) for MI455X / gfx1250, bf16 WMMA + fp32 accum.
// B=8, S=512, D=1024, L=3S=1536.
// Round 2: async global->LDS tile copies (ASYNCcnt path) for bf16 GEMM stages.
// ---------------------------------------------------------------------------

#define BSZ 8
#define SEG 512
#define DIM 1024
#define LEN 1536            // 3*SEG
#define NEGV (-1.0e8f)

#define BLK 128             // block tile M=N=128
#define BK  32              // k-step per WMMA (bf16 16x16x32)
#define LDT 40              // padded LDS row stride (elements): 80 bytes, 16B aligned

typedef __attribute__((ext_vector_type(16))) __bf16 bf16x16;
typedef __attribute__((ext_vector_type(8)))  __bf16 bf16x8;
typedef __attribute__((ext_vector_type(8)))  float  f32x8;
typedef __attribute__((ext_vector_type(4)))  float  f32x4;

// A fragment (16x32 bf16, MxK): lanes 0-15 -> K 0..7 & 16..23, lanes 16-31 -> K 8..15 & 24..31
__device__ __forceinline__ bf16x16 frag_a(const __bf16* tile, int row, int lane) {
    int kb = (lane >> 4) * 8;
    const __bf16* p = tile + row * LDT + kb;
    union { bf16x16 v; bf16x8 h[2]; } u;
    u.h[0] = *(const bf16x8*)p;
    u.h[1] = *(const bf16x8*)(p + 16);
    return u.v;
}

// B fragment (32x16 bf16, KxN), tile stored n-major / k-contiguous:
// lanes 0-15 -> K 0..7 & 16..23 of column (lane&15), lanes 16-31 -> K 8..15 & 24..31
__device__ __forceinline__ bf16x16 frag_b(const __bf16* tile, int col, int lane) {
    int kb = (lane >> 4) * 8;
    const __bf16* p = tile + col * LDT + kb;
    union { bf16x16 v; bf16x8 h[2]; } u;
    u.h[0] = *(const bf16x8*)p;
    u.h[1] = *(const bf16x8*)(p + 16);
    return u.v;
}

__device__ __forceinline__ f32x8 wmma_bf16(bf16x16 a, bf16x16 b, f32x8 c) {
    return __builtin_amdgcn_wmma_f32_16x16x32_bf16(false, a, false, b, (short)0, c,
                                                   false, false);
}

// Async 16-byte global -> LDS copy (no VGPR staging), tracked by ASYNCcnt.
// lds_off: raw LDS byte offset (low 32 bits of the flat shared address).
__device__ __forceinline__ void async_cp16(unsigned lds_off, const __bf16* g) {
    asm volatile("global_load_async_to_lds_b128 %0, %1, off"
                 :: "v"(lds_off), "v"(g)
                 : "memory");
}
__device__ __forceinline__ void wait_async0() {
    asm volatile("s_wait_asynccnt 0x0" ::: "memory");
}

// ---------------------------------------------------------------------------
// Kernel 1: fused Q/K/V projections.  grid.z = 9 -> (which = z/3, seg g = z%3)
// Y[g] = H[g] @ W[which][g] + b[which][g]   ->  bf16 [B, 3S, D] (concat layout)
// Keeps the VGPR path: fp32 source must pass through v_cvt_pk_bf16_f32 anyway.
// ---------------------------------------------------------------------------
__global__ __launch_bounds__(256)
void proj_kernel(const float* __restrict__ h1, const float* __restrict__ h2,
                 const float* __restrict__ h3,
                 const float* __restrict__ Wq, const float* __restrict__ bq,
                 const float* __restrict__ Wk, const float* __restrict__ bk,
                 const float* __restrict__ Wv, const float* __restrict__ bv,
                 __bf16* __restrict__ Qb, __bf16* __restrict__ Kb,
                 __bf16* __restrict__ Vb) {
    const int mat   = blockIdx.z;
    const int g     = mat % 3;
    const int which = mat / 3;
    const float* H    = (g == 0) ? h1 : (g == 1) ? h2 : h3;
    const float* W    = ((which == 0) ? Wq : (which == 1) ? Wk : Wv) + (size_t)g * DIM * DIM;
    const float* bias = ((which == 0) ? bq : (which == 1) ? bk : bv) + (size_t)g * DIM;
    __bf16* dst = (which == 0) ? Qb : (which == 1) ? Kb : Vb;

    const int m0 = blockIdx.y * BLK;   // rows in [0, B*S)
    const int n0 = blockIdx.x * BLK;   // cols in [0, D)

    __shared__ __bf16 As[BLK * LDT];
    __shared__ __bf16 Bs[BLK * LDT];

    const int tid = threadIdx.x, lane = tid & 31, wid = tid >> 5;
    const int wm = (wid & 3) * 32, wn = (wid >> 2) * 64;

    f32x8 acc[2][4] = {};

    for (int kk = 0; kk < DIM; kk += BK) {
        // ---- A tile: H rows, fp32 -> bf16 on the fly
        {
            const int r = tid >> 1, hh = (tid & 1) * 16;
            const float* src = H + (size_t)(m0 + r) * DIM + kk + hh;
            f32x4 t[4];
            t[0] = *(const f32x4*)(src);
            t[1] = *(const f32x4*)(src + 4);
            t[2] = *(const f32x4*)(src + 8);
            t[3] = *(const f32x4*)(src + 12);
            __bf16* d = As + r * LDT + hh;
#pragma unroll
            for (int j = 0; j < 4; ++j)
#pragma unroll
                for (int e = 0; e < 4; ++e) d[j * 4 + e] = (__bf16)t[j][e];
            if (kk + BK < DIM) __builtin_prefetch(src + BK, 0, 1);
        }
        // ---- B tile: W (k-major in memory) stored transposed -> Bs[n][k]
        {
            const int kr = tid >> 3, nc = (tid & 7) * 16;
            const float* src = W + (size_t)(kk + kr) * DIM + n0 + nc;
            f32x4 t[4];
            t[0] = *(const f32x4*)(src);
            t[1] = *(const f32x4*)(src + 4);
            t[2] = *(const f32x4*)(src + 8);
            t[3] = *(const f32x4*)(src + 12);
#pragma unroll
            for (int j = 0; j < 4; ++j)
#pragma unroll
                for (int e = 0; e < 4; ++e)
                    Bs[(nc + j * 4 + e) * LDT + kr] = (__bf16)t[j][e];
            if (kk + BK < DIM) __builtin_prefetch(src + (size_t)BK * DIM, 0, 1);
        }
        __syncthreads();

        bf16x16 bfrag[4];
#pragma unroll
        for (int nt = 0; nt < 4; ++nt) bfrag[nt] = frag_b(Bs, wn + nt * 16 + (lane & 15), lane);
#pragma unroll
        for (int mt = 0; mt < 2; ++mt) {
            bf16x16 a = frag_a(As, wm + mt * 16 + (lane & 15), lane);
#pragma unroll
            for (int nt = 0; nt < 4; ++nt) acc[mt][nt] = wmma_bf16(a, bfrag[nt], acc[mt][nt]);
        }
        __syncthreads();
    }

    // epilogue: +bias, write bf16 to concatenated [B, 3S, D]
    const int lr = (lane >> 4) * 8, lc = lane & 15;
#pragma unroll
    for (int nt = 0; nt < 4; ++nt) {
        const int col = n0 + wn + nt * 16 + lc;
        const float bv_ = bias[col];
#pragma unroll
        for (int mt = 0; mt < 2; ++mt)
#pragma unroll
            for (int r = 0; r < 8; ++r) {
                const int row = m0 + wm + mt * 16 + lr + r;
                const int b = row >> 9, s = row & (SEG - 1);
                dst[((size_t)b * LEN + g * SEG + s) * DIM + col] =
                    (__bf16)(acc[mt][nt][r] + bv_);
            }
    }
}

// ---------------------------------------------------------------------------
// Kernel 2: scores = (Q @ K^T) * sqrt(D) + key_mask + block_exclusion  (fp32)
// A and B tiles are bf16 & layout-natural -> async global->LDS copies.
// ---------------------------------------------------------------------------
__global__ __launch_bounds__(256)
void scores_kernel(const __bf16* __restrict__ Qb, const __bf16* __restrict__ Kb,
                   const int* __restrict__ m1, const int* __restrict__ m2,
                   const int* __restrict__ m3, float* __restrict__ Sc) {
    const int b  = blockIdx.z;
    const int m0 = blockIdx.y * BLK;   // query rows
    const int n0 = blockIdx.x * BLK;   // key cols
    const __bf16* Q = Qb + (size_t)b * LEN * DIM;
    const __bf16* K = Kb + (size_t)b * LEN * DIM;
    float* S = Sc + (size_t)b * LEN * LEN;

    __shared__ __bf16 As[BLK * LDT];
    __shared__ __bf16 Bs[BLK * LDT];

    const int tid = threadIdx.x, lane = tid & 31, wid = tid >> 5;
    const int wm = (wid & 3) * 32, wn = (wid >> 2) * 64;

    const int r  = tid >> 1, hh = (tid & 1) * 16;
    const unsigned la = (unsigned)(uintptr_t)(As + r * LDT + hh);
    const unsigned lb = (unsigned)(uintptr_t)(Bs + r * LDT + hh);

    f32x8 acc[2][4] = {};

    for (int kk = 0; kk < DIM; kk += BK) {
        const __bf16* ga = Q + (size_t)(m0 + r) * DIM + kk + hh;   // A: Q rows
        const __bf16* gb = K + (size_t)(n0 + r) * DIM + kk + hh;   // B: K rows (n-major)
        async_cp16(la,      ga);
        async_cp16(la + 16, ga + 8);
        async_cp16(lb,      gb);
        async_cp16(lb + 16, gb + 8);
        wait_async0();
        __syncthreads();

        bf16x16 bfrag[4];
#pragma unroll
        for (int nt = 0; nt < 4; ++nt) bfrag[nt] = frag_b(Bs, wn + nt * 16 + (lane & 15), lane);
#pragma unroll
        for (int mt = 0; mt < 2; ++mt) {
            bf16x16 a = frag_a(As, wm + mt * 16 + (lane & 15), lane);
#pragma unroll
            for (int nt = 0; nt < 4; ++nt) acc[mt][nt] = wmma_bf16(a, bfrag[nt], acc[mt][nt]);
        }
        __syncthreads();
    }

    const int lr = (lane >> 4) * 8, lc = lane & 15;
    const int segq = m0 >> 9;                 // whole 128-row block is one segment
#pragma unroll
    for (int nt = 0; nt < 4; ++nt) {
        const int col = n0 + wn + nt * 16 + lc;
        const int segk = col >> 9;
        const int* mp = (segk == 0) ? m1 : (segk == 1) ? m2 : m3;
        float addv = (mp[b * SEG + (col & (SEG - 1))] == 1) ? 0.0f : NEGV;
        if (segk == segq) addv += NEGV;       // same-segment block exclusion
#pragma unroll
        for (int mt = 0; mt < 2; ++mt)
#pragma unroll
            for (int rr = 0; rr < 8; ++rr) {
                const int row = m0 + wm + mt * 16 + lr + rr;
                S[(size_t)row * LEN + col] = acc[mt][nt][rr] * 32.0f + addv;
            }
    }
}

// ---------------------------------------------------------------------------
// Kernel 3: row softmax over 1536, output P as bf16.  1 workgroup / row.
// ---------------------------------------------------------------------------
__global__ __launch_bounds__(256)
void softmax_kernel(const float* __restrict__ Sc, __bf16* __restrict__ P) {
    const int rid = blockIdx.x;               // [0, B*LEN)
    const float* x = Sc + (size_t)rid * LEN;
    __bf16* p = P + (size_t)rid * LEN;
    __shared__ float red[256];
    const int tid = threadIdx.x;

    float t[6];
#pragma unroll
    for (int j = 0; j < 6; ++j) t[j] = x[tid + j * 256];

    float mx = t[0];
#pragma unroll
    for (int j = 1; j < 6; ++j) mx = fmaxf(mx, t[j]);
    red[tid] = mx;
    __syncthreads();
    for (int off = 128; off > 0; off >>= 1) {
        if (tid < off) red[tid] = fmaxf(red[tid], red[tid + off]);
        __syncthreads();
    }
    mx = red[0];
    __syncthreads();

    float sum = 0.0f;
#pragma unroll
    for (int j = 0; j < 6; ++j) { t[j] = __expf(t[j] - mx); sum += t[j]; }
    red[tid] = sum;
    __syncthreads();
    for (int off = 128; off > 0; off >>= 1) {
        if (tid < off) red[tid] += red[tid + off];
        __syncthreads();
    }
    const float inv = 1.0f / red[0];
#pragma unroll
    for (int j = 0; j < 6; ++j) p[tid + j * 256] = (__bf16)(t[j] * inv);
}

// ---------------------------------------------------------------------------
// Kernel 4: O = P @ V + V   (fp32 out, pre-LayerNorm)
// A tile async; B tile needs transpose -> VGPR staging + transposed ds stores.
// ---------------------------------------------------------------------------
__global__ __launch_bounds__(256)
void pv_kernel(const __bf16* __restrict__ P, const __bf16* __restrict__ Vb,
               float* __restrict__ O) {
    const int b  = blockIdx.z;
    const int m0 = blockIdx.y * BLK;   // query rows
    const int n0 = blockIdx.x * BLK;   // d cols
    const __bf16* Pp = P  + (size_t)b * LEN * LEN;
    const __bf16* V  = Vb + (size_t)b * LEN * DIM;
    float* Ob = O + (size_t)b * LEN * DIM;

    __shared__ __bf16 As[BLK * LDT];
    __shared__ __bf16 Bs[BLK * LDT];

    const int tid = threadIdx.x, lane = tid & 31, wid = tid >> 5;
    const int wm = (wid & 3) * 32, wn = (wid >> 2) * 64;

    const int r  = tid >> 1, hh = (tid & 1) * 16;
    const unsigned la = (unsigned)(uintptr_t)(As + r * LDT + hh);

    f32x8 acc[2][4] = {};

    for (int kk = 0; kk < LEN; kk += BK) {
        // A tile: P rows (k = key index, contiguous) -> async copy
        const __bf16* ga = Pp + (size_t)(m0 + r) * LEN + kk + hh;
        async_cp16(la,      ga);
        async_cp16(la + 16, ga + 8);

        {   // B tile: V[key][d] -> transpose on LDS write: Bs[d][key]
            const int kr = tid >> 3, nc = (tid & 7) * 16;
            const __bf16* src = V + (size_t)(kk + kr) * DIM + n0 + nc;
            union { bf16x8 v[2]; __bf16 e[16]; } t;
            t.v[0] = ((const bf16x8*)src)[0];
            t.v[1] = ((const bf16x8*)src)[1];
#pragma unroll
            for (int j = 0; j < 16; ++j) Bs[(nc + j) * LDT + kr] = t.e[j];
            if (kk + BK < LEN) __builtin_prefetch(src + (size_t)BK * DIM, 0, 1);
        }
        wait_async0();
        __syncthreads();

        bf16x16 bfrag[4];
#pragma unroll
        for (int nt = 0; nt < 4; ++nt) bfrag[nt] = frag_b(Bs, wn + nt * 16 + (lane & 15), lane);
#pragma unroll
        for (int mt = 0; mt < 2; ++mt) {
            bf16x16 a = frag_a(As, wm + mt * 16 + (lane & 15), lane);
#pragma unroll
            for (int nt = 0; nt < 4; ++nt) acc[mt][nt] = wmma_bf16(a, bfrag[nt], acc[mt][nt]);
        }
        __syncthreads();
    }

    const int lr = (lane >> 4) * 8, lc = lane & 15;
#pragma unroll
    for (int nt = 0; nt < 4; ++nt) {
        const int col = n0 + wn + nt * 16 + lc;
#pragma unroll
        for (int mt = 0; mt < 2; ++mt)
#pragma unroll
            for (int rr = 0; rr < 8; ++rr) {
                const int row = m0 + wm + mt * 16 + lr + rr;
                const float res = (float)V[(size_t)row * DIM + col];   // residual +V
                Ob[(size_t)row * DIM + col] = acc[mt][nt][rr] + res;
            }
    }
}

// ---------------------------------------------------------------------------
// Kernel 5: per-row LayerNorm with per-segment affine; scatter to tuple layout
// out[g][b][s][d], g = q/S.
// ---------------------------------------------------------------------------
__global__ __launch_bounds__(256)
void ln_kernel(const float* __restrict__ O, const float* __restrict__ lng,
               const float* __restrict__ lnb, float* __restrict__ out) {
    const int rid = blockIdx.x;                 // b*LEN + q
    const int b = rid / LEN, q = rid % LEN;
    const int g = q >> 9, s = q & (SEG - 1);
    const float* x  = O + (size_t)rid * DIM;
    const float* gg = lng + (size_t)g * DIM;
    const float* bb = lnb + (size_t)g * DIM;
    float* y = out + (((size_t)g * BSZ + b) * SEG + s) * DIM;

    __shared__ float red[256];
    const int tid = threadIdx.x;

    float t[4];
#pragma unroll
    for (int j = 0; j < 4; ++j) t[j] = x[tid + j * 256];

    float sum = t[0] + t[1] + t[2] + t[3];
    red[tid] = sum;
    __syncthreads();
    for (int off = 128; off > 0; off >>= 1) {
        if (tid < off) red[tid] += red[tid + off];
        __syncthreads();
    }
    const float mu = red[0] * (1.0f / DIM);
    __syncthreads();

    float sq = 0.0f;
#pragma unroll
    for (int j = 0; j < 4; ++j) { const float d = t[j] - mu; sq += d * d; }
    red[tid] = sq;
    __syncthreads();
    for (int off = 128; off > 0; off >>= 1) {
        if (tid < off) red[tid] += red[tid + off];
        __syncthreads();
    }
    const float inv = rsqrtf(red[0] * (1.0f / DIM) + 1e-8f);
#pragma unroll
    for (int j = 0; j < 4; ++j) {
        const int c = tid + j * 256;
        y[c] = (t[j] - mu) * inv * gg[c] + bb[c];
    }
}

// ---------------------------------------------------------------------------
extern "C" void kernel_launch(void* const* d_in, const int* in_sizes, int n_in,
                              void* d_out, int out_size, void* d_ws, size_t ws_size,
                              hipStream_t stream) {
    const float* h1 = (const float*)d_in[0];
    const float* h2 = (const float*)d_in[1];
    const float* h3 = (const float*)d_in[2];
    const float* Wq = (const float*)d_in[3];
    const float* bq = (const float*)d_in[4];
    const float* Wk = (const float*)d_in[5];
    const float* bk = (const float*)d_in[6];
    const float* Wv = (const float*)d_in[7];
    const float* bv = (const float*)d_in[8];
    const float* lng = (const float*)d_in[9];
    const float* lnb = (const float*)d_in[10];
    const int* m1 = (const int*)d_in[11];
    const int* m2 = (const int*)d_in[12];
    const int* m3 = (const int*)d_in[13];
    float* out = (float*)d_out;

    // workspace layout
    char* ws = (char*)d_ws;
    const size_t qkv = (size_t)BSZ * LEN * DIM;   // elements
    const size_t ll  = (size_t)BSZ * LEN * LEN;
    __bf16* Qb = (__bf16*)ws;
    __bf16* Kb = Qb + qkv;
    __bf16* Vb = Kb + qkv;
    float*  Sc = (float*)(Vb + qkv);
    __bf16* P  = (__bf16*)(Sc + ll);
    float*  O  = (float*)(P + ll);

    // 1) projections: 9 GEMMs [4096x1024]x[1024x1024]
    proj_kernel<<<dim3(DIM / BLK, (BSZ * SEG) / BLK, 9), 256, 0, stream>>>(
        h1, h2, h3, Wq, bq, Wk, bk, Wv, bv, Qb, Kb, Vb);
    // 2) masked scores
    scores_kernel<<<dim3(LEN / BLK, LEN / BLK, BSZ), 256, 0, stream>>>(
        Qb, Kb, m1, m2, m3, Sc);
    // 3) softmax -> bf16 probs
    softmax_kernel<<<dim3(BSZ * LEN), 256, 0, stream>>>(Sc, P);
    // 4) P @ V + V
    pv_kernel<<<dim3(DIM / BLK, LEN / BLK, BSZ), 256, 0, stream>>>(P, Vb, O);
    // 5) LayerNorm + scatter to (o1, o2, o3)
    ln_kernel<<<dim3(BSZ * LEN), 256, 0, stream>>>(O, lng, lnb, out);
}
